// GeoGATConv_31894427140227
// MI455X (gfx1250) — matrix operand order repaired and checked by reading, not compiled, not run on hardware
//
#include <hip/hip_runtime.h>
#include <cstdint>

typedef __attribute__((ext_vector_type(16))) _Float16 v16h;
typedef __attribute__((ext_vector_type(8)))  _Float16 v8h;
typedef __attribute__((ext_vector_type(8)))  float    v8f;

#define NEG_SLOPE 0.2f
#define NHEADS 3

// ---------- monotone float <-> uint encoding for atomic float max ----------
__device__ __forceinline__ unsigned fenc(float f) {
    unsigned u = __float_as_uint(f);
    return (u & 0x80000000u) ? ~u : (u | 0x80000000u);
}
__device__ __forceinline__ float fdec(unsigned u) {
    unsigned b = (u & 0x80000000u) ? (u & 0x7FFFFFFFu) : ~u;
    return __uint_as_float(b);
}
#define ENC_NEG_INF 0x007FFFFFu   // fenc(-inf)

__device__ __forceinline__ v16h mk16(v8h lo, v8h hi) {
    v16h r;
#pragma unroll
    for (int i = 0; i < 8; ++i) { r[i] = lo[i]; r[i + 8] = hi[i]; }
    return r;
}

// ---------------------------- fill ----------------------------
__global__ void fill_u32(unsigned* __restrict__ p, unsigned v, int n) {
    int i = blockIdx.x * blockDim.x + threadIdx.x;
    if (i < n) p[i] = v;
}

// ---------------------------- conversions / transposes ----------------------------
__global__ void f32_to_f16_k(const float* __restrict__ in, _Float16* __restrict__ out, int n) {
    int i = blockIdx.x * blockDim.x + threadIdx.x;
    if (i < n) out[i] = (_Float16)in[i];
}
// W1 [256,192] f32 -> W1T [192,256] f16
__global__ void transpose_w1(const float* __restrict__ W, _Float16* __restrict__ WT) {
    int t = blockIdx.x * blockDim.x + threadIdx.x;
    if (t >= 256 * 192) return;
    int k = t / 192, c = t % 192;
    WT[c * 256 + k] = (_Float16)W[t];
}
// W2 [64,30] f32 -> W2T [32,64] f16 (cols 30,31 zero-padded)
__global__ void transpose_w2(const float* __restrict__ W, _Float16* __restrict__ WT) {
    int t = blockIdx.x * blockDim.x + threadIdx.x;
    if (t >= 32 * 64) return;
    int c = t / 64, k = t % 64;
    WT[t] = (_Float16)((c < 30) ? W[k * 30 + c] : 0.0f);
}

// ---------------------------- GEMM1: f16[N,256] x f16T[192,256] -> f32[N,192] ----------------------------
// One wave computes a 16x64 output tile (4 accumulators); block = 3 waves -> full 192 cols.
__global__ __launch_bounds__(96) void gemm1_wmma(const _Float16* __restrict__ A,
                                                 const _Float16* __restrict__ BT,
                                                 float* __restrict__ out, int N) {
    const int K = 256, NB = 192;
    const int wave = threadIdx.x >> 5;
    const int lane = threadIdx.x & 31;
    const int rowBase = blockIdx.x * 16;
    const int colBase = wave * 64;
    const int r  = lane & 15;
    const int hi = lane >> 4;            // 0 or 1
    int rr = rowBase + r; if (rr >= N) rr = N - 1;
    const _Float16* Arow = A + (size_t)rr * K;
    const int koffA = hi * 8;            // A: K in {koffA..+7} U {16+koffA..+7}
    const int koffB = hi * 8;            // B: K in {koffB..+7} U {16+koffB..+7}
    const int n = lane & 15;

    v8f c[4] = {v8f{}, v8f{}, v8f{}, v8f{}};
    for (int k0 = 0; k0 < K; k0 += 32) {
        v16h a = mk16(*(const v8h*)(Arow + k0 + koffA),
                      *(const v8h*)(Arow + k0 + 16 + koffA));
#pragma unroll
        for (int j = 0; j < 4; ++j) {
            const _Float16* Bp = BT + (size_t)(colBase + j * 16 + n) * K + k0 + koffB;
            v16h b = mk16(*(const v8h*)Bp, *(const v8h*)(Bp + 16));
            c[j] = __builtin_amdgcn_wmma_f32_16x16x32_f16(
                       false, a, false, b, (short)0, c[j], false, false);
        }
    }
    const int mb = hi * 8;
#pragma unroll
    for (int j2 = 0; j2 < 8; ++j2) {
        int row = rowBase + mb + j2;
        if (row < N) {
            float* op = out + (size_t)row * NB + colBase + n;
            op[0]  = c[0][j2];
            op[16] = c[1][j2];
            op[32] = c[2][j2];
            op[48] = c[3][j2];
        }
    }
}

// ---------------------------- GEMM2: f16[N,64] x f16T[32,64] -> f32[N,30] ----------------------------
__global__ __launch_bounds__(128) void gemm2_wmma(const _Float16* __restrict__ A,
                                                  const _Float16* __restrict__ BT,
                                                  float* __restrict__ out, int N) {
    const int K = 64, NB = 30;
    const int wave = threadIdx.x >> 5;
    const int lane = threadIdx.x & 31;
    const int rowBase = (blockIdx.x * 4 + wave) * 16;
    if (rowBase >= N) return;
    const int r  = lane & 15;
    const int hi = lane >> 4;
    int rr = rowBase + r; if (rr >= N) rr = N - 1;
    const _Float16* Arow = A + (size_t)rr * K;
    const int koffA = hi * 8;
    const int koffB = hi * 8;
    const int n = lane & 15;

    v8f c[2] = {v8f{}, v8f{}};
#pragma unroll
    for (int k0 = 0; k0 < K; k0 += 32) {
        v16h a = mk16(*(const v8h*)(Arow + k0 + koffA),
                      *(const v8h*)(Arow + k0 + 16 + koffA));
#pragma unroll
        for (int j = 0; j < 2; ++j) {
            const _Float16* Bp = BT + (size_t)(j * 16 + n) * K + k0 + koffB;
            v16h b = mk16(*(const v8h*)Bp, *(const v8h*)(Bp + 16));
            c[j] = __builtin_amdgcn_wmma_f32_16x16x32_f16(
                       false, a, false, b, (short)0, c[j], false, false);
        }
    }
    const int mb = hi * 8;
#pragma unroll
    for (int j2 = 0; j2 < 8; ++j2) {
        int row = rowBase + mb + j2;
        if (row < N) {
#pragma unroll
            for (int j = 0; j < 2; ++j) {
                int col = j * 16 + n;
                if (col < NB) out[(size_t)row * NB + col] = c[j][j2];
            }
        }
    }
}

// ---------------------------- per-node attention dots ----------------------------
template <int C>
__global__ void node_attn(const float* __restrict__ xh,
                          const float* __restrict__ att_s,
                          const float* __restrict__ att_d,
                          float* __restrict__ as_, float* __restrict__ ad_, int N) {
    int t = blockIdx.x * blockDim.x + threadIdx.x;   // t = n*3 + h
    if (t >= N * NHEADS) return;
    int h = t % NHEADS, nn = t / NHEADS;
    const float* row = xh + (size_t)nn * (NHEADS * C) + h * C;
    float s = 0.f, d = 0.f;
#pragma unroll 4
    for (int c = 0; c < C; ++c) {
        s += row[c] * att_s[h * C + c];
        d += row[c] * att_d[h * C + c];
    }
    as_[t] = s;
    ad_[t] = d;
}

// ---------------------------- edge pass 1: segment max ----------------------------
__global__ void edge_max(const int* __restrict__ ei, int E, int N,
                         const float* __restrict__ as_, const float* __restrict__ ad_,
                         unsigned* __restrict__ menc) {
    int e = blockIdx.x * blockDim.x + threadIdx.x;
    int E2 = E + N;
    if (e >= E2) return;
    int s = (e < E) ? ei[e]     : (e - E);
    int d = (e < E) ? ei[E + e] : (e - E);
#pragma unroll
    for (int h = 0; h < NHEADS; ++h) {
        float v = as_[s * NHEADS + h] + ad_[d * NHEADS + h];
        v = (v > 0.f) ? v : NEG_SLOPE * v;
        atomicMax(&menc[d * NHEADS + h], fenc(v));
    }
}

// ---------------------------- edge pass 2: exp + segment sum ----------------------------
__global__ void edge_expsum(const int* __restrict__ ei, int E, int N,
                            const float* __restrict__ as_, const float* __restrict__ ad_,
                            const unsigned* __restrict__ menc,
                            float* __restrict__ exbuf, float* __restrict__ ssum) {
    int e = blockIdx.x * blockDim.x + threadIdx.x;
    int E2 = E + N;
    if (e >= E2) return;
    int s = (e < E) ? ei[e]     : (e - E);
    int d = (e < E) ? ei[E + e] : (e - E);
#pragma unroll
    for (int h = 0; h < NHEADS; ++h) {
        float v = as_[s * NHEADS + h] + ad_[d * NHEADS + h];
        v = (v > 0.f) ? v : NEG_SLOPE * v;
        float m  = fdec(menc[d * NHEADS + h]);
        float ex = expf(v - m);
        exbuf[(size_t)e * NHEADS + h] = ex;
        atomicAdd(&ssum[d * NHEADS + h], ex);
    }
}

// ---------------------------- edge pass 3: weighted message scatter ----------------------------
template <int C>
__global__ void edge_scatter(const int* __restrict__ ei, int E, int N,
                             const float* __restrict__ xh,
                             const float* __restrict__ exbuf,
                             const float* __restrict__ ssum,
                             float* __restrict__ acc) {
    int tid = blockIdx.x * blockDim.x + threadIdx.x;
    int E2 = E + N;
    int e = tid / C, c = tid % C;
    if (e >= E2) return;
    int s = (e < E) ? ei[e]     : (e - E);
    int d = (e < E) ? ei[E + e] : (e - E);
#pragma unroll
    for (int h = 0; h < NHEADS; ++h) {
        float alpha = exbuf[(size_t)e * NHEADS + h] / ssum[d * NHEADS + h];
        float msg = xh[(size_t)s * (NHEADS * C) + h * C + c] * alpha;
        atomicAdd(&acc[(size_t)d * (NHEADS * C) + h * C + c], msg);
    }
}

// ---------------------------- finalize layer 1: mean heads + bias + relu -> f16 ----------------------------
__global__ void finalize1(const float* __restrict__ acc, const float* __restrict__ b1,
                          _Float16* __restrict__ x2h, int N) {
    int t = blockIdx.x * blockDim.x + threadIdx.x;   // t = n*64 + c
    if (t >= N * 64) return;
    int nn = t >> 6, c = t & 63;
    const float* a = acc + (size_t)nn * 192;
    float v = (a[c] + a[64 + c] + a[128 + c]) * (1.0f / 3.0f) + b1[c];
    x2h[t] = (_Float16)fmaxf(v, 0.0f);
}

// ---------------------------- finalize layer 2: mean heads + bias + log_softmax ----------------------------
__global__ void finalize2(const float* __restrict__ acc, const float* __restrict__ b2,
                          float* __restrict__ out, int N) {
    int nn = blockIdx.x * blockDim.x + threadIdx.x;
    if (nn >= N) return;
    const float* a = acc + (size_t)nn * 30;
    float z[10];
    float m = -3.4e38f;
#pragma unroll
    for (int c = 0; c < 10; ++c) {
        z[c] = (a[c] + a[10 + c] + a[20 + c]) * (1.0f / 3.0f) + b2[c];
        m = fmaxf(m, z[c]);
    }
    float s = 0.f;
#pragma unroll
    for (int c = 0; c < 10; ++c) s += expf(z[c] - m);
    float lse = m + logf(s);
#pragma unroll
    for (int c = 0; c < 10; ++c) out[(size_t)nn * 10 + c] = z[c] - lse;
}

// ---------------------------- launch ----------------------------
extern "C" void kernel_launch(void* const* d_in, const int* in_sizes, int n_in,
                              void* d_out, int out_size, void* d_ws, size_t ws_size,
                              hipStream_t stream) {
    const float* feat  = (const float*)d_in[0];
    const int*   ei    = (const int*)d_in[1];
    const float* W1    = (const float*)d_in[2];
    const float* atts1 = (const float*)d_in[3];
    const float* attd1 = (const float*)d_in[4];
    const float* b1    = (const float*)d_in[5];
    const float* W2    = (const float*)d_in[6];
    const float* atts2 = (const float*)d_in[7];
    const float* attd2 = (const float*)d_in[8];
    const float* b2    = (const float*)d_in[9];
    float* out = (float*)d_out;

    const int N  = in_sizes[0] / 256;
    const int E  = in_sizes[1] / 2;
    const int E2 = E + N;

    // ---------- workspace layout (float units, all chunks multiple-of-4 floats => 16B aligned) ----------
    auto up4 = [](size_t x) { return (x + 3) & ~(size_t)3; };
    float* wsf = (float*)d_ws;
    const size_t N192 = (size_t)N * 192;
    float*    xh1  = wsf;                               // N*192
    float*    acc1 = xh1 + N192;                        // N*192 (featH parks here before acc1 is needed)
    _Float16* featH = (_Float16*)acc1;                  // N*256 f16 == N*128 floats (fits in acc1)
    float*    as1  = acc1 + N192;                       // N*3
    float*    ad1  = as1 + up4(3 * (size_t)N);          // N*3
    float*    s1   = ad1 + up4(3 * (size_t)N);          // N*3
    unsigned* m1   = (unsigned*)(s1 + up4(3 * (size_t)N)); // N*3
    float*    ex1  = (float*)m1 + up4(3 * (size_t)N);   // E2*3
    _Float16* x2h  = (_Float16*)(ex1 + up4((size_t)E2 * 3)); // N*64 f16 == N*32 floats
    _Float16* W1T  = x2h + (size_t)N * 64;              // 192*256 f16
    _Float16* W2T  = W1T + 192 * 256;                   // 32*64 f16
    // layer-2 buffers reuse the xh1 region (xh1 dead after edge_scatter<64>)
    float*    xh2  = xh1;                               // N*30
    float*    as2  = xh2 + up4((size_t)N * 30);         // N*3
    float*    ad2  = as2 + up4(3 * (size_t)N);          // N*3
    float*    s2   = ad2 + up4(3 * (size_t)N);          // N*3
    unsigned* m2   = (unsigned*)(s2 + up4(3 * (size_t)N)); // N*3
    float*    ex2  = (float*)m2 + up4(3 * (size_t)N);   // E2*3
    float*    acc2 = ex2 + up4((size_t)E2 * 3);         // N*30

    const int TB = 256;
    auto blk = [](long long n, int b) { return (int)((n + b - 1) / b); };

    // ===== precision staging (f16 operands for WMMA) =====
    f32_to_f16_k<<<blk((long long)N * 256, TB), TB, 0, stream>>>(feat, featH, N * 256);
    transpose_w1<<<blk(256LL * 192, TB), TB, 0, stream>>>(W1, W1T);
    transpose_w2<<<blk(32LL * 64, TB), TB, 0, stream>>>(W2, W2T);

    // ===== layer 1 =====
    gemm1_wmma<<<(N + 15) / 16, 96, 0, stream>>>(featH, W1T, xh1, N);
    // acc1 zero-fill AFTER gemm1 (acc1 region hosted featH until now)
    fill_u32<<<blk((long long)N * 192, TB), TB, 0, stream>>>((unsigned*)acc1, 0u, N * 192);
    fill_u32<<<blk(3LL * N, TB), TB, 0, stream>>>((unsigned*)s1, 0u, 3 * N);
    fill_u32<<<blk(3LL * N, TB), TB, 0, stream>>>(m1, ENC_NEG_INF, 3 * N);

    node_attn<64><<<blk(3LL * N, TB), TB, 0, stream>>>(xh1, atts1, attd1, as1, ad1, N);
    edge_max<<<blk(E2, TB), TB, 0, stream>>>(ei, E, N, as1, ad1, m1);
    edge_expsum<<<blk(E2, TB), TB, 0, stream>>>(ei, E, N, as1, ad1, m1, ex1, s1);
    edge_scatter<64><<<blk((long long)E2 * 64, TB), TB, 0, stream>>>(ei, E, N, xh1, ex1, s1, acc1);
    finalize1<<<blk((long long)N * 64, TB), TB, 0, stream>>>(acc1, b1, x2h, N);

    // ===== layer 2 =====
    fill_u32<<<blk(30LL * N, TB), TB, 0, stream>>>((unsigned*)acc2, 0u, 30 * N);
    fill_u32<<<blk(3LL * N, TB), TB, 0, stream>>>((unsigned*)s2, 0u, 3 * N);
    fill_u32<<<blk(3LL * N, TB), TB, 0, stream>>>(m2, ENC_NEG_INF, 3 * N);

    gemm2_wmma<<<((N + 15) / 16 + 3) / 4, 128, 0, stream>>>(x2h, W2T, xh2, N);
    node_attn<10><<<blk(3LL * N, TB), TB, 0, stream>>>(xh2, atts2, attd2, as2, ad2, N);
    edge_max<<<blk(E2, TB), TB, 0, stream>>>(ei, E, N, as2, ad2, m2);
    edge_expsum<<<blk(E2, TB), TB, 0, stream>>>(ei, E, N, as2, ad2, m2, ex2, s2);
    edge_scatter<10><<<blk((long long)E2 * 10, TB), TB, 0, stream>>>(ei, E, N, xh2, ex2, s2, acc2);
    finalize2<<<blk(N, 128), 128, 0, stream>>>(acc2, b2, out, N);
}
